// Model_67705864454686
// MI455X (gfx1250) — matrix operand order repaired and checked by reading, not compiled, not run on hardware
//
#include <hip/hip_runtime.h>
#include <hip/hip_bf16.h>

// ---------------------------------------------------------------------------
// MI455X (gfx1250) implementation.
//  - GEMM-shaped stages -> v_wmma_f32_16x16x32_bf16 (wave32, 16x16 tiles,
//    4 N-tiles per wave for A-fragment reuse and back-to-back WMMA issue)
//  - A and B both pre-packed to bf16 fragment-linear layout: inner loop is
//    5 contiguous 32B loads (1 A + 4 B) + 4 wmma, zero divergence.
//  - FFT convolution -> 32768-pt complex FFT fully inside LDS (256KB/WGP),
//    frequency-domain channel reduction (one inverse FFT total).
//  - d_ws usage ~175MB.
// ---------------------------------------------------------------------------

#define LEAKF 0.2f
#define PI_F  3.14159265358979323846f
#define NF    32768      // complex FFT length (real length 65536)
#define NSAMP 32768
#define ATOMN 16384

typedef __attribute__((ext_vector_type(8)))  float  v8f;
typedef __attribute__((ext_vector_type(16))) __bf16 v16bf;

// ===========================================================================
// Weight packing: f32 -> bf16 fragments in the 16x32 A layout
// (ISA 7.12.2: lane&15 = M, element e -> K = e + (e>=8)*8 + (lane>=16)*8)
// Packed layout: [tap][M/16][K/32][32 lanes][16 elems]
// ===========================================================================
__global__ void pack_w_kernel(const float* __restrict__ W, __bf16* __restrict__ A,
                              int M, int K, int ntaps, int t0, int t1, int t2,
                              int srcTaps, int transposed, long total) {
  long idx = (long)blockIdx.x * 256 + threadIdx.x;
  if (idx >= total) return;
  int  e    = (int)(idx & 15);
  int  lane = (int)((idx >> 4) & 31);
  long v    = idx >> 9;
  int  ktiles = K >> 5;
  int  kt = (int)(v % ktiles); v /= ktiles;
  int  mtiles = M >> 4;
  int  mt = (int)(v % mtiles);
  int  tap = (int)(v / mtiles);
  int  m  = (mt << 4) + (lane & 15);
  int  kk = e + ((e >= 8) ? 8 : 0) + ((lane & 16) ? 8 : 0);
  int  k  = (kt << 5) + kk;
  int  ksel = (tap == 0) ? t0 : ((tap == 1) ? t1 : t2);
  long src = transposed ? (((long)k * M + m) * srcTaps + ksel)
                        : (((long)m * K + k) * srcTaps + ksel);
  A[idx] = (__bf16)W[src];
}

// ===========================================================================
// Activation packing: f32 [K][xld] -> bf16 B fragments with tap shift and
// zero padding baked in.  B layout: lane&15 = N,
// element e -> K = e + (e>=8)*8 + (lane>=16)*8.
// Packed layout: [tap][K/32][N/16][32 lanes][16 elems]
// ===========================================================================
__global__ void pack_b_kernel(const float* __restrict__ X, __bf16* __restrict__ B,
                              int Kdim, int Nlen, int xld, int nvalid,
                              int s0, int s1, int s2, long total) {
  long idx = (long)blockIdx.x * 256 + threadIdx.x;
  if (idx >= total) return;
  int  e    = (int)(idx & 15);
  int  lane = (int)((idx >> 4) & 31);
  long v    = idx >> 9;
  int  ntiles = Nlen >> 4;
  int  nt = (int)(v % ntiles); v /= ntiles;
  int  ktiles = Kdim >> 5;
  int  kt = (int)(v % ktiles);
  int  tap = (int)(v / ktiles);
  int  n = (nt << 4) + (lane & 15);
  int  k = (kt << 5) + e + ((e >= 8) ? 8 : 0) + ((lane & 16) ? 8 : 0);
  int  sh = (tap == 0) ? s0 : ((tap == 1) ? s1 : s2);
  int  t = n + sh;
  float xv = (t >= 0 && t < nvalid) ? X[(size_t)k * xld + t] : 0.0f;
  B[idx] = (__bf16)xv;
}

// ===========================================================================
// WMMA GEMM on packed fragments. One wave = 1 M-tile x 4 N-tiles.
// Inner loop: 1 A-fragment load (reused 4x) + 4 B-fragment loads + 4 wmma.
// ===========================================================================
__global__ void __launch_bounds__(256) wmma_gemm_kernel(
    const __bf16* __restrict__ Apack, const __bf16* __restrict__ Bpack,
    float* __restrict__ Y, const float* __restrict__ bias,
    int mtiles, int ktiles, int ntiles, int ntaps,
    long yld, int ystride, int yoff, int act) {
  int wave = threadIdx.x >> 5;
  int lane = threadIdx.x & 31;
  int ngroups = ntiles >> 2;
  int grp = blockIdx.x * 8 + wave;
  if (grp >= mtiles * ngroups) return;
  int mt  = grp / ngroups;
  int nt0 = (grp - mt * ngroups) << 2;

  int lm = lane & 15;
  int hi = (lane >> 4) & 1;

  v8f acc[4];
#pragma unroll
  for (int j = 0; j < 4; ++j)
#pragma unroll
    for (int r = 0; r < 8; ++r) acc[j][r] = 0.0f;

  for (int tap = 0; tap < ntaps; ++tap) {
    const __bf16* Ab = Apack + (((size_t)tap * mtiles + mt) * ktiles) * 512 + lane * 16;
    const __bf16* Bb = Bpack + (((size_t)tap * ktiles) * ntiles + nt0) * 512 + lane * 16;
    for (int kt = 0; kt < ktiles; ++kt) {
      v16bf a = *(const v16bf*)(Ab + (size_t)kt * 512);
      const __bf16* bb = Bb + (size_t)kt * ntiles * 512;
      v16bf b0 = *(const v16bf*)(bb);
      v16bf b1 = *(const v16bf*)(bb + 512);
      v16bf b2 = *(const v16bf*)(bb + 1024);
      v16bf b3 = *(const v16bf*)(bb + 1536);
      acc[0] = __builtin_amdgcn_wmma_f32_16x16x32_bf16(false, a, false, b0, (short)0, acc[0], false, false);
      acc[1] = __builtin_amdgcn_wmma_f32_16x16x32_bf16(false, a, false, b1, (short)0, acc[1], false, false);
      acc[2] = __builtin_amdgcn_wmma_f32_16x16x32_bf16(false, a, false, b2, (short)0, acc[2], false, false);
      acc[3] = __builtin_amdgcn_wmma_f32_16x16x32_bf16(false, a, false, b3, (short)0, acc[3], false, false);
    }
  }
#pragma unroll
  for (int j = 0; j < 4; ++j) {
    int n = ((nt0 + j) << 4) + lm;
#pragma unroll
    for (int r = 0; r < 8; ++r) {                // C/D layout: lanes16-31 -> M+8
      int m = (mt << 4) + r + (hi << 3);
      float v = acc[j][r] + bias[m];
      if (act == 1) v = v > 0.f ? v : LEAKF * v;
      else if (act == 2) v = v > 0.f ? v : 0.f;
      Y[(size_t)m * yld + (size_t)n * ystride + yoff] = v;
    }
  }
}

// ===========================================================================
// Embed: h[e*128+a][b] = x[a][b][:] @ embed_w[:,e] + embed_b[e]
// ===========================================================================
__global__ void embed_kernel(const float* __restrict__ x, const float* __restrict__ W,
                             const float* __restrict__ bias, float* __restrict__ h) {
  int idx = blockIdx.x * 256 + threadIdx.x;     // 131072
  int e = idx & 7, ab = idx >> 3;
  int a = ab >> 7, b = ab & 127;
  const float* xr = x + (size_t)(a * 128 + b) * 257;
  float acc = bias[e];
  for (int k = 0; k < 257; ++k) acc += xr[k] * W[k * 8 + e];
  h[(size_t)(e * 128 + a) * 128 + b] = acc;
}

// BatchNorm over length L per channel (biased var), optional trailing leaky.
__global__ void bn_kernel(float* __restrict__ Y, const float* __restrict__ g,
                          const float* __restrict__ b, int L, int post_leaky) {
  int ch = blockIdx.x;
  float* row = Y + (size_t)ch * L;
  __shared__ float s1[256], s2[256];
  float a = 0.f, q = 0.f;
  for (int t = threadIdx.x; t < L; t += 256) { float v = row[t]; a += v; q += v * v; }
  s1[threadIdx.x] = a; s2[threadIdx.x] = q; __syncthreads();
  for (int o = 128; o > 0; o >>= 1) {
    if (threadIdx.x < o) { s1[threadIdx.x] += s1[threadIdx.x + o]; s2[threadIdx.x] += s2[threadIdx.x + o]; }
    __syncthreads();
  }
  float mu = s1[0] / (float)L;
  float var = s2[0] / (float)L - mu * mu;
  float inv = rsqrtf(var + 1e-5f) * g[ch];
  float bb = b[ch];
  for (int t = threadIdx.x; t < L; t += 256) {
    float v = (row[t] - mu) * inv + bb;
    if (post_leaky) v = v > 0.f ? v : LEAKF * v;
    row[t] = v;
  }
}

__global__ void rowsum_kernel(const float* __restrict__ X, float* __restrict__ out, int L) {
  __shared__ float s1[256];
  const float* row = X + (size_t)blockIdx.x * L;
  float a = 0.f;
  for (int t = threadIdx.x; t < L; t += 256) a += row[t];
  s1[threadIdx.x] = a; __syncthreads();
  for (int o = 128; o > 0; o >>= 1) { if (threadIdx.x < o) s1[threadIdx.x] += s1[threadIdx.x + o]; __syncthreads(); }
  if (threadIdx.x == 0) out[blockIdx.x] = s1[0];
}

// ===========================================================================
// Atom bank: per-frame hann-window L2 norms, then fused window/norm/OLA/resonance.
// ===========================================================================
__global__ void frame_norm_kernel(const float* __restrict__ atoms, float* __restrict__ norms) {
  int fr = blockIdx.x;                        // 1024*63
  int ch = fr / 63, f = fr - ch * 63;
  const float* base = atoms + (size_t)ch * ATOMN + f * 256;
  __shared__ float s1[256];
  float a = 0.f;
  for (int s = threadIdx.x; s < 512; s += 256) {
    float w = 0.5f - 0.5f * __cosf(2.0f * PI_F * (float)s / 512.0f);
    float v = base[s] * w;
    a += v * v;
  }
  s1[threadIdx.x] = a; __syncthreads();
  for (int o = 128; o > 0; o >>= 1) { if (threadIdx.x < o) s1[threadIdx.x] += s1[threadIdx.x + o]; __syncthreads(); }
  if (threadIdx.x == 0) norms[fr] = sqrtf(s1[0]);
}

__global__ void atoms_build_kernel(const float* __restrict__ atomsIn, const float* __restrict__ norms,
                                   const float* __restrict__ resonance, float* __restrict__ outp) {
  size_t idx = (size_t)blockIdx.x * 256 + threadIdx.x;   // 1024*16384
  int ch = (int)(idx >> 14);
  int t  = (int)(idx & (ATOMN - 1));
  float x = atomsIn[idx];
  float acc = 0.f;
  int f1 = t >> 8;
  if (f1 <= 62) {
    int s = t - (f1 << 8);
    float w = 0.5f - 0.5f * __cosf(2.0f * PI_F * (float)s / 512.0f);
    acc += w / (norms[ch * 63 + f1] + 1e-8f);
  }
  int f0 = f1 - 1;
  if (f0 >= 0) {
    int s = t - (f0 << 8);
    float w = 0.5f - 0.5f * __cosf(2.0f * PI_F * (float)s / 512.0f);
    acc += w / (norms[ch * 63 + f0] + 1e-8f);
  }
  float r = fminf(fmaxf(resonance[ch], 0.1f), 1.0f);     // cumprod+linear interp of r^j
  float pos = ((float)t + 0.5f) / 256.0f - 0.5f;
  int i0 = (int)floorf(pos); i0 = i0 < 0 ? 0 : (i0 > 63 ? 63 : i0);
  int i1 = i0 + 1; if (i1 > 63) i1 = 63;
  float w = fminf(fmaxf(pos - (float)i0, 0.f), 1.f);
  float res = powf(r, (float)(i0 + 1)) * (1.f - w) + powf(r, (float)(i1 + 1)) * w;
  outp[idx] = x * acc * res;
}

// ===========================================================================
// In-LDS 32768-pt complex FFT (bit-reverse + DIT), 1024 threads, 256KB LDS.
// sgn = -1 forward, +1 inverse (unnormalized).
// ===========================================================================
__device__ __forceinline__ void fft_lds(float2* sm, int tid, int nth, float sgn) {
  for (int i = tid; i < NF; i += nth) {
    int j = (int)(__brev((unsigned)i) >> 17);           // 32-15 bits
    if (j > i) { float2 a = sm[i]; sm[i] = sm[j]; sm[j] = a; }
  }
  __syncthreads();
  for (int s = 1; s <= 15; ++s) {
    int mh = 1 << (s - 1);
    for (int b = tid; b < (NF >> 1); b += nth) {
      int pos = b & (mh - 1);
      int i0 = ((b >> (s - 1)) << s) + pos;
      int i1 = i0 + mh;
      float ang = sgn * PI_F * (float)pos / (float)mh;
      float sn, cs; __sincosf(ang, &sn, &cs);
      float2 u = sm[i0], t = sm[i1];
      float tr = t.x * cs - t.y * sn;
      float ti = t.x * sn + t.y * cs;
      sm[i0] = make_float2(u.x + tr, u.y + ti);
      sm[i1] = make_float2(u.x - tr, u.y - ti);
    }
    __syncthreads();
  }
}

// rfft(65536 real) of one row (zero-padded past nvalid), split & store X[0..NF].
__global__ void __launch_bounds__(1024) fft_fwd_store_kernel(
    const float* __restrict__ src, long rowstride, int nvalid, float2* __restrict__ out) {
  extern __shared__ float2 sm[];
  int tid = threadIdx.x, nth = blockDim.x;
  const float* row = src + (size_t)blockIdx.x * rowstride;
  for (int i = tid; i < NF; i += nth) {
    int n0 = 2 * i, n1 = 2 * i + 1;
    float a = (n0 < nvalid) ? row[n0] : 0.f;
    float b = (n1 < nvalid) ? row[n1] : 0.f;
    sm[i] = make_float2(a, b);
  }
  __syncthreads();
  fft_lds(sm, tid, nth, -1.0f);
  float2* o = out + (size_t)blockIdx.x * (NF + 1);
  for (int k = tid; k <= NF; k += nth) {
    int k1 = k & (NF - 1), k2 = (NF - k) & (NF - 1);
    float2 Zk = sm[k1], Zn = sm[k2];
    float2 Xe = make_float2(0.5f * (Zk.x + Zn.x), 0.5f * (Zk.y - Zn.y));
    float2 Xo = make_float2(0.5f * (Zk.y + Zn.y), 0.5f * (Zn.x - Zk.x));
    float ang = -PI_F * (float)k / (float)NF;
    float sn, cs; __sincosf(ang, &sn, &cs);
    o[k] = make_float2(Xe.x + cs * Xo.x - sn * Xo.y, Xe.y + cs * Xo.y + sn * Xo.x);
  }
}

// g = noise*encoded (fused), rfft, multiply with stashed atom spectrum,
// atomically accumulate product spectrum over channels.
__global__ void __launch_bounds__(1024) fft_g_mulacc_kernel(
    const float* __restrict__ noise, const float* __restrict__ enc,
    const float2* __restrict__ Xa, float* __restrict__ Sacc) {
  extern __shared__ float2 sm[];
  int tid = threadIdx.x, nth = blockDim.x;
  const float* nr = noise + (size_t)blockIdx.x * NSAMP;
  const float* er = enc + (size_t)blockIdx.x * NSAMP;
  for (int i = tid; i < NF; i += nth) {
    int n0 = 2 * i, n1 = 2 * i + 1;
    float a = (n0 < NSAMP) ? nr[n0] * er[n0] : 0.f;
    float b = (n1 < NSAMP) ? nr[n1] * er[n1] : 0.f;
    sm[i] = make_float2(a, b);
  }
  __syncthreads();
  fft_lds(sm, tid, nth, -1.0f);
  const float2* xa = Xa + (size_t)blockIdx.x * (NF + 1);
  for (int k = tid; k <= NF; k += nth) {
    int k1 = k & (NF - 1), k2 = (NF - k) & (NF - 1);
    float2 Zk = sm[k1], Zn = sm[k2];
    float2 Xe = make_float2(0.5f * (Zk.x + Zn.x), 0.5f * (Zk.y - Zn.y));
    float2 Xo = make_float2(0.5f * (Zk.y + Zn.y), 0.5f * (Zn.x - Zk.x));
    float ang = -PI_F * (float)k / (float)NF;
    float sn, cs; __sincosf(ang, &sn, &cs);
    float2 Xg = make_float2(Xe.x + cs * Xo.x - sn * Xo.y, Xe.y + cs * Xo.y + sn * Xo.x);
    float2 A = xa[k];
    atomicAdd(&Sacc[2 * k],     A.x * Xg.x - A.y * Xg.y);
    atomicAdd(&Sacc[2 * k + 1], A.x * Xg.y + A.y * Xg.x);
  }
}

// irfft(65536) from spectrum S[0..NF] -> 65536 real samples.
__global__ void __launch_bounds__(1024) fft_inv_kernel(
    const float2* __restrict__ S, float* __restrict__ outt) {
  extern __shared__ float2 sm[];
  int tid = threadIdx.x, nth = blockDim.x;
  for (int k = tid; k < NF; k += nth) {
    float2 Xk = S[k], Xn = S[NF - k];
    float2 Xe = make_float2(0.5f * (Xk.x + Xn.x), 0.5f * (Xk.y - Xn.y));
    float2 D  = make_float2(0.5f * (Xk.x - Xn.x), 0.5f * (Xk.y + Xn.y));
    float ang = PI_F * (float)k / (float)NF;
    float sn, cs; __sincosf(ang, &sn, &cs);
    float2 Xo = make_float2(cs * D.x - sn * D.y, cs * D.y + sn * D.x);
    sm[k] = make_float2(Xe.x - Xo.y, Xe.y + Xo.x);      // Z' = Xe + i*Xo
  }
  __syncthreads();
  fft_lds(sm, tid, nth, 1.0f);
  const float scale = 1.0f / (float)NF;
  for (int i = tid; i < NF; i += nth) {
    outt[2 * i]     = sm[i].x * scale;
    outt[2 * i + 1] = sm[i].y * scale;
  }
}

__global__ void spec_mul_kernel(const float2* __restrict__ A, const float2* __restrict__ B,
                                float2* __restrict__ C) {
  int k = blockIdx.x * 256 + threadIdx.x;
  if (k > NF) return;
  float2 a = A[k], b = B[k];
  C[k] = make_float2(a.x * b.x - a.y * b.y, a.x * b.y + a.y * b.x);
}

// ===========================================================================
// Reverb head: 3x(1024x1024 matvec + leaky), layernorm, room/mix softmax, impulse.
// ===========================================================================
__global__ void __launch_bounds__(1024) reverb_kernel(
    const float* __restrict__ ctx,
    const float* __restrict__ w0, const float* __restrict__ b0,
    const float* __restrict__ w1, const float* __restrict__ b1,
    const float* __restrict__ w2, const float* __restrict__ b2,
    const float* __restrict__ lng, const float* __restrict__ lnb,
    const float* __restrict__ to_rooms, const float* __restrict__ rooms,
    const float* __restrict__ to_mix,
    float* __restrict__ impulse, float* __restrict__ mixout) {
  __shared__ float c[1024];
  __shared__ float red[1024];
  __shared__ float sc[2];
  __shared__ float rw[8];
  int i = threadIdx.x;
  c[i] = ctx[i];
  __syncthreads();
  const float* Ws[3] = {w0, w1, w2};
  const float* Bs[3] = {b0, b1, b2};
  for (int l = 0; l < 3; ++l) {
    const float* W = Ws[l];
    float acc = Bs[l][i];
    for (int j = 0; j < 1024; ++j) acc += c[j] * W[j * 1024 + i];
    acc = acc > 0.f ? acc : LEAKF * acc;
    __syncthreads();
    c[i] = acc;
    __syncthreads();
  }
  red[i] = c[i]; __syncthreads();
  for (int o = 512; o > 0; o >>= 1) { if (i < o) red[i] += red[i + o]; __syncthreads(); }
  if (i == 0) sc[0] = red[0] * (1.0f / 1024.0f);
  __syncthreads();
  float mu = sc[0];
  float d = c[i] - mu;
  red[i] = d * d; __syncthreads();
  for (int o = 512; o > 0; o >>= 1) { if (i < o) red[i] += red[i + o]; __syncthreads(); }
  if (i == 0) sc[1] = red[0] * (1.0f / 1024.0f);
  __syncthreads();
  float cn = d * rsqrtf(sc[1] + 1e-5f) * lng[i] + lnb[i];
  __syncthreads();
  c[i] = cn;
  __syncthreads();
  if (i < 8) {
    float a = 0.f;
    for (int j = 0; j < 1024; ++j) a += c[j] * to_rooms[j * 8 + i];
    rw[i] = a;
  }
  __syncthreads();
  if (i == 0) {
    float mx = rw[0];
    for (int j = 1; j < 8; ++j) mx = fmaxf(mx, rw[j]);
    float s = 0.f;
    for (int j = 0; j < 8; ++j) { rw[j] = __expf(rw[j] - mx); s += rw[j]; }
    for (int j = 0; j < 8; ++j) rw[j] /= s;
  }
  __syncthreads();
  for (int t = i; t < NSAMP; t += 1024) {
    float a = 0.f;
#pragma unroll
    for (int j = 0; j < 8; ++j) a += rw[j] * rooms[(size_t)j * NSAMP + t];
    impulse[t] = a;
  }
  if (i < 2) {
    float a = 0.f;
    for (int j = 0; j < 1024; ++j) a += c[j] * to_mix[j * 2 + i];
    red[i] = a;
  }
  __syncthreads();
  if (i == 0) {
    float mx = fmaxf(red[0], red[1]);
    float e0 = __expf(red[0] - mx), e1 = __expf(red[1] - mx);
    mixout[0] = e0 / (e0 + e1);
    mixout[1] = e1 / (e0 + e1);
  }
}

__global__ void final_mix_kernel(const float* __restrict__ dry, const float* __restrict__ wet,
                                 const float* __restrict__ mix, float* __restrict__ out) {
  int t = blockIdx.x * 256 + threadIdx.x;
  out[t] = dry[t] * mix[0] + wet[t] * mix[1];
}

// ===========================================================================
// Host orchestration
// ===========================================================================
extern "C" void kernel_launch(void* const* d_in, const int* in_sizes, int n_in,
                              void* d_out, int out_size, void* d_ws, size_t ws_size,
                              hipStream_t stream) {
  (void)in_sizes; (void)n_in; (void)out_size; (void)ws_size;
  auto F = [&](int i) { return (const float*)d_in[i]; };
  // input map (setup_inputs flat order):
  // 0 x, 1 noise, 2 iteration, 3 embed_w, 4 embed_b, 5-8 ctx_conv_w, 9-12 ctx_conv_b,
  // 13-16 ctx_bn_g, 17-20 ctx_bn_b, 21 sal_w, 22 sal_b, 23 up_w0, 24 up_b0,
  // 25-32 ups_w, 33-40 ups_b, 41-48 ups_bn_g, 49-56 ups_bn_b, 57 up_final_w,
  // 58 up_final_b, 59 resonance, 60 atoms, 61-63 verb_net_w, 64-66 verb_net_b,
  // 67 verb_ln_g, 68 verb_ln_b, 69 verb_rooms, 70 verb_to_rooms, 71 verb_to_mix
  float* out = (float*)d_out;
  float* encoded = out + NSAMP;               // output tuple: [out | encoded]

  char* wsb = (char*)d_ws;
  size_t off = 0;
  auto alloc = [&](size_t bytes) -> char* {
    char* p = wsb + off;
    off += (bytes + 511) & ~(size_t)511;
    return p;
  };
  float*  H0       = (float*)alloc((size_t)1024 * 128 * 4);
  float*  H1       = (float*)alloc((size_t)1024 * 128 * 4);
  float*  UPA      = (float*)alloc((size_t)128 * NSAMP * 4);
  float*  UPB      = (float*)alloc((size_t)128 * NSAMP * 4);
  float*  ctxsum   = (float*)alloc(1024 * 4);
  float*  norms    = (float*)alloc((size_t)1024 * 63 * 4);
  float*  atomsBuf = (float*)alloc((size_t)1024 * ATOMN * 4);
  float2* sacc     = (float2*)alloc((size_t)(NF + 1) * 8);
  float2* stash    = (float2*)alloc((size_t)64 * (NF + 1) * 8);
  float2* xd       = (float2*)alloc((size_t)(NF + 1) * 8);
  float2* xi       = (float2*)alloc((size_t)(NF + 1) * 8);
  float2* s2       = (float2*)alloc((size_t)(NF + 1) * 8);
  float*  dryFull  = (float*)alloc((size_t)2 * NF * 4);
  float*  wetFull  = (float*)alloc((size_t)2 * NF * 4);
  float*  impulse  = (float*)alloc((size_t)NSAMP * 4);
  float*  mix      = (float*)alloc(64);
  __bf16* pk       = (__bf16*)alloc(14680064ull * 2);     // packed bf16 weights
  __bf16* bp       = (__bf16*)alloc(12582912ull * 2);     // packed bf16 activations (reused)

  __bf16* ctxW[4];
  for (int i = 0; i < 4; ++i) ctxW[i] = pk + (size_t)i * 3 * 1024 * 1024;
  __bf16* salW = pk + 4ull * 3 * 1024 * 1024;
  __bf16* up0W = salW + 1024ull * 1024;
  __bf16* upsW = up0W + 128ull * 1024;                    // [8][2] x (2*128*128)
  __bf16* upfW = upsW + 16ull * 2 * 128 * 128;

  auto pack = [&](const float* W, __bf16* A, int M, int K, int ntaps,
                  int t0, int t1, int t2, int srcTaps, int transposed) {
    long total = (long)ntaps * M * K;
    pack_w_kernel<<<(unsigned)((total + 255) / 256), 256, 0, stream>>>(
        W, A, M, K, ntaps, t0, t1, t2, srcTaps, transposed, total);
  };
  // pack activations + run WMMA GEMM
  auto gemm = [&](const __bf16* Ap, const float* X, float* Y, const float* bias,
                  int M, int K, int Nlen, int xld, int nvalid,
                  int s0, int s1, int s2, int ntaps,
                  long yld, int ystride, int yoff, int act) {
    long btot = (long)ntaps * K * Nlen;
    pack_b_kernel<<<(unsigned)((btot + 255) / 256), 256, 0, stream>>>(
        X, bp, K, Nlen, xld, nvalid, s0, s1, s2, btot);
    int mtiles = M >> 4, ktiles = K >> 5, ntiles = Nlen >> 4;
    int groups = mtiles * (ntiles >> 2);
    wmma_gemm_kernel<<<(groups + 7) / 8, 256, 0, stream>>>(
        Ap, bp, Y, bias, mtiles, ktiles, ntiles, ntaps, yld, ystride, yoff, act);
  };

  // ---- pack all weights to bf16 fragment layout (once per call) ----
  for (int i = 0; i < 4; ++i) pack(F(5 + i), ctxW[i], 1024, 1024, 3, 0, 1, 2, 3, 0);
  pack(F(21), salW, 1024, 1024, 1, 0, 0, 0, 1, 0);
  pack(F(23), up0W, 128, 1024, 1, 0, 0, 0, 1, 0);
  for (int l = 0; l < 8; ++l) {                    // convT even phase: taps k=1,3; odd: k=0,2
    pack(F(25 + l), upsW + (size_t)(l * 2 + 0) * 2 * 128 * 128, 128, 128, 2, 1, 3, 0, 4, 1);
    pack(F(25 + l), upsW + (size_t)(l * 2 + 1) * 2 * 128 * 128, 128, 128, 2, 0, 2, 0, 4, 1);
  }
  pack(F(57), upfW, 1024, 128, 3, 0, 1, 2, 3, 0);

  // ---- embed + dilated context stack (WMMA) ----
  embed_kernel<<<512, 256, 0, stream>>>(F(0), F(3), F(4), H0);
  int dil[4] = {1, 3, 9, 1};
  float* pa = H0; float* pb = H1;
  for (int i = 0; i < 4; ++i) {
    gemm(ctxW[i], pa, pb, F(9 + i), 1024, 1024, 128, 128, 128,
         -dil[i], 0, dil[i], 3, 128, 1, 0, /*leaky*/1);
    bn_kernel<<<1024, 256, 0, stream>>>(pb, F(13 + i), F(17 + i), 128, 0);
    float* t = pa; pa = pb; pb = t;
  }
  gemm(salW, pa, pb, F(22), 1024, 1024, 128, 128, 128, 0, 0, 0, 1, 128, 1, 0, 0);
  gemm(up0W, pb, UPA, F(24), 128, 1024, 128, 128, 128, 0, 0, 0, 1, 128, 1, 0, 0);

  // ---- learned upsampler 128 -> 32768 (WMMA, even/odd strided phases) ----
  float* cur = UPA; float* nxt = UPB;
  int Li = 128;
  for (int l = 0; l < 8; ++l) {
    const __bf16* We = upsW + (size_t)(l * 2 + 0) * 2 * 128 * 128;
    const __bf16* Wo = upsW + (size_t)(l * 2 + 1) * 2 * 128 * 128;
    gemm(We, cur, nxt, F(33 + l), 128, 128, Li, Li, Li, 0, -1, 0, 2, 2L * Li, 2, 0, 0);
    gemm(Wo, cur, nxt, F(33 + l), 128, 128, Li, Li, Li, 1,  0, 0, 2, 2L * Li, 2, 1, 0);
    bn_kernel<<<128, 256, 0, stream>>>(nxt, F(41 + l), F(49 + l), 2 * Li, /*post leaky*/1);
    float* t = cur; cur = nxt; nxt = t;
    Li *= 2;
  }
  // up_final 128->1024 K=3 + ReLU -> encoded (second tuple output), ~1.57M wmma issues
  gemm(upfW, cur, encoded, F(58), 1024, 128, NSAMP, NSAMP, NSAMP,
       -1, 0, 1, 3, NSAMP, 1, 0, /*relu*/2);

  // ---- ctx = sum(encoded), reverb head ----
  rowsum_kernel<<<1024, 256, 0, stream>>>(encoded, ctxsum, NSAMP);
  reverb_kernel<<<1, 1024, 0, stream>>>(ctxsum, F(61), F(64), F(62), F(65), F(63), F(66),
                                        F(67), F(68), F(70), F(69), F(71), impulse, mix);

  // ---- atom bank (window/norm/OLA/resonance fused) ----
  frame_norm_kernel<<<1024 * 63, 256, 0, stream>>>(F(60), norms);
  atoms_build_kernel<<<(1024 * ATOMN) / 256, 256, 0, stream>>>(F(60), norms, F(59), atomsBuf);

  // ---- frequency-domain channel reduction: sum_ch Fa_ch * Fg_ch ----
  const size_t SH = (size_t)NF * sizeof(float2);   // 256KB LDS per workgroup
  hipMemsetAsync(sacc, 0, (size_t)(NF + 1) * sizeof(float2), stream);
  for (int c0 = 0; c0 < 1024; c0 += 64) {
    fft_fwd_store_kernel<<<64, 1024, SH, stream>>>(atomsBuf + (size_t)c0 * ATOMN,
                                                   (long)ATOMN, ATOMN, stash);
    fft_g_mulacc_kernel<<<64, 1024, SH, stream>>>(F(1) + (size_t)c0 * NSAMP,
                                                  encoded + (size_t)c0 * NSAMP,
                                                  stash, (float*)sacc);
  }
  fft_inv_kernel<<<1, 1024, SH, stream>>>(sacc, dryFull);   // dry = first 32768

  // ---- reverb convolution + mix ----
  fft_fwd_store_kernel<<<1, 1024, SH, stream>>>(dryFull, (long)NSAMP, NSAMP, xd);
  fft_fwd_store_kernel<<<1, 1024, SH, stream>>>(impulse, (long)NSAMP, NSAMP, xi);
  spec_mul_kernel<<<(NF + 1 + 255) / 256, 256, 0, stream>>>(xd, xi, s2);
  fft_inv_kernel<<<1, 1024, SH, stream>>>(s2, wetFull);
  final_mix_kernel<<<NSAMP / 256, 256, 0, stream>>>(dryFull, wetFull, mix, out);
}
